// BorderDet_11982958756176
// MI455X (gfx1250) — matrix operand order repaired and checked by reading, not compiled, not run on hardware
//
#include <hip/hip_runtime.h>
#include <stdint.h>

// ---------------------------------------------------------------------------
// BorderDet head for MI455X (gfx1250, wave32, WMMA).
// All heavy math runs through one bf16 WMMA GEMM (V_WMMA_F32_16X16X32_BF16,
// fp32 accumulation). Convs are implicit GEMMs via im2col (tap-major K);
// norms / border align are light VALU kernels.
// GEMM: block = 128 thr = 4 waves; each wave owns a 64x64 tile (4x4 WMMA
// accumulators, 128 VGPRs) -> 16 WMMAs per 16 b128 loads, no branches in the
// k-loop (N-tile count is a template parameter).
// ---------------------------------------------------------------------------

typedef __attribute__((ext_vector_type(16))) __bf16 v16bf;
typedef __attribute__((ext_vector_type(8)))  float  v8f;

#define NBATCH 2
#define CDIM   256
#define HDIM   64
#define WDIM   64
#define HWDIM  4096
#define MROWS  8192      // NBATCH*HWDIM
#define KCONV  2304      // CDIM*9

__device__ __forceinline__ unsigned short f2bf(float f) {
  union { float f; unsigned u; } c; c.f = f;
  unsigned u = c.u;
  u += 0x7fffu + ((u >> 16) & 1u);      // round-to-nearest-even
  return (unsigned short)(u >> 16);
}

union Frag { uint4 q[2]; v16bf v; };

// ---------------------------------------------------------------------------
// WMMA GEMM: D[M x N] = A[M x K] * B^T   (B stored as N rows of K, row-major)
// A, B bf16 (as ushort), D fp32. M fixed = 8192.
// Wave tile 64 x (NT*16); block tile 256 x (NT*16); grid.y = 32.
// ---------------------------------------------------------------------------
template <int NT>
__global__ void __launch_bounds__(128)
k_gemm_bf16(const unsigned short* __restrict__ A,
            const unsigned short* __restrict__ B,
            float* __restrict__ D, int Ndim, int Kdim, int nOff)
{
  const int wave = threadIdx.x >> 5;
  const int lane = threadIdx.x & 31;
  const int l    = lane & 15;
  const int hi   = lane >> 4;
  const int rowBase = blockIdx.y * 256 + wave * 64;
  const int nBase   = blockIdx.x * (NT * 16) + nOff;

  const v8f zero = {0.f,0.f,0.f,0.f,0.f,0.f,0.f,0.f};
  v8f acc[4][NT];
#pragma unroll
  for (int mt = 0; mt < 4; ++mt)
#pragma unroll
    for (int nt = 0; nt < NT; ++nt) acc[mt][nt] = zero;

  // A fragment: lane<16 holds row rowBase+l, K {0..7, 16..23}; lane>=16 K {8..15, 24..31}
  const unsigned short* pa[4];
#pragma unroll
  for (int mt = 0; mt < 4; ++mt)
    pa[mt] = A + (size_t)(rowBase + mt * 16 + l) * Kdim + hi * 8;
  // B fragment: lane<16 -> col nBase+nt*16+l holds K {0..7, 16..23} ; lane>=16 -> K {8..15, 24..31}
  const unsigned short* pb[NT];
#pragma unroll
  for (int nt = 0; nt < NT; ++nt)
    pb[nt] = B + (size_t)(nBase + nt * 16 + l) * Kdim + hi * 8;

  for (int k0 = 0; k0 < Kdim; k0 += 32) {
    Frag a[4], b[NT];
#pragma unroll
    for (int mt = 0; mt < 4; ++mt) {
      a[mt].q[0] = *(const uint4*)(pa[mt] + k0);
      a[mt].q[1] = *(const uint4*)(pa[mt] + k0 + 16);
    }
#pragma unroll
    for (int nt = 0; nt < NT; ++nt) {
      b[nt].q[0] = *(const uint4*)(pb[nt] + k0);
      b[nt].q[1] = *(const uint4*)(pb[nt] + k0 + 16);
    }
#pragma unroll
    for (int nt = 0; nt < NT; ++nt)
#pragma unroll
      for (int mt = 0; mt < 4; ++mt)
        acc[mt][nt] = __builtin_amdgcn_wmma_f32_16x16x32_bf16(
            false, a[mt].v, false, b[nt].v, (short)0, acc[mt][nt], false, false);
  }

#pragma unroll
  for (int mt = 0; mt < 4; ++mt) {
    const int rb = rowBase + mt * 16 + hi * 8;
#pragma unroll
    for (int nt = 0; nt < NT; ++nt) {
      const int col = nBase + nt * 16 + l;
      float* dp = D + (size_t)rb * Ndim + col;
#pragma unroll
      for (int vi = 0; vi < 8; ++vi)
        dp[(size_t)vi * Ndim] = acc[mt][nt][vi];
    }
  }
}

// ---------------------------------------------------------------------------
// Layout / cast helpers
// ---------------------------------------------------------------------------
__global__ void k_nchw2nhwc(const float* __restrict__ x, float* __restrict__ out)
{
  int idx = blockIdx.x * 256 + threadIdx.x;          // over N*C*HW (coalesced read)
  int p = idx & 4095;
  int c = (idx >> 12) & 255;
  int n = idx >> 20;
  out[((size_t)((n << 12) | p)) * 256 + c] = x[idx];
}

// act NHWC fp32 [M][256] -> col bf16 [M][2304], tap-major k = t*256 + c.
// Per tap, 256 consecutive threads write 512 contiguous bytes (coalesced).
__global__ void k_im2col(const float* __restrict__ act, unsigned short* __restrict__ col)
{
  int idx = blockIdx.x * 256 + threadIdx.x;          // over M*256
  int c = idx & 255;
  int m = idx >> 8;
  int n = m >> 12, p = m & 4095, h = p >> 6, w = p & 63;
  unsigned short* o = col + (size_t)m * KCONV + c;
#pragma unroll
  for (int ky = 0; ky < 3; ++ky) {
    int hh = h + ky - 1;
#pragma unroll
    for (int kx = 0; kx < 3; ++kx) {
      int wl = w + kx - 1;
      float v = 0.f;
      if ((unsigned)hh < 64u && (unsigned)wl < 64u)
        v = act[((size_t)((n << 12) + (hh << 6) + wl)) * 256 + c];
      o[(ky * 3 + kx) * 256] = f2bf(v);
    }
  }
}

// 3x3 conv weights (rows x 256 x 3 x 3, OIHW) -> bf16 [rows][2304] tap-major:
// dst[co][t*256 + c] = src[co][c*9 + t]
__global__ void k_conv_weight(const float* __restrict__ src, unsigned short* __restrict__ dst,
                              int total /* rows*2304 */)
{
  int idx = blockIdx.x * 256 + threadIdx.x;
  if (idx >= total) return;
  int co = idx / KCONV, k = idx - co * KCONV;
  int t = k >> 8, c = k & 255;
  dst[idx] = f2bf(src[(size_t)co * KCONV + c * 9 + t]);
}

// fp32 -> bf16, zero-fill padded region [real, total)  (1x1 conv weights etc.)
__global__ void k_cast_pad(const float* __restrict__ src, unsigned short* __restrict__ dst,
                           int real, int total)
{
  int idx = blockIdx.x * 256 + threadIdx.x;
  if (idx >= total) return;
  dst[idx] = (idx < real) ? f2bf(src[idx]) : (unsigned short)0;
}

// bbox_pred_w (4x2304) + ctr_w (1x2304) -> 16x2304 bf16 tap-major (rows 5..15 zero)
__global__ void k_pack_head(const float* __restrict__ bbw, const float* __restrict__ ctw,
                            unsigned short* __restrict__ dst)
{
  int idx = blockIdx.x * 256 + threadIdx.x;          // over 16*2304
  if (idx >= 16 * KCONV) return;
  int r = idx / KCONV, k = idx - r * KCONV;
  int t = k >> 8, c = k & 255;
  int ks = c * 9 + t;
  float v = (r < 4) ? bbw[(size_t)r * KCONV + ks] : ((r == 4) ? ctw[ks] : 0.f);
  dst[idx] = f2bf(v);
}

// ---------------------------------------------------------------------------
// GroupNorm (32 groups of 8 chans) over NHWC [M][256]; bias added pre-norm
// ---------------------------------------------------------------------------
__global__ void k_gn_stats(const float* __restrict__ Y, const float* __restrict__ bias,
                           float* __restrict__ mean, float* __restrict__ rstd)
{
  __shared__ float ssum[256], ssq[256];
  int b = blockIdx.x;                 // 64 = N*32
  int n = b >> 5, g = b & 31;
  const float* bb = bias + g * 8;
  float s = 0.f, sq = 0.f;
  for (int p = threadIdx.x; p < HWDIM; p += 256) {
    const float* row = Y + ((size_t)(n * HWDIM + p)) * 256 + g * 8;
#pragma unroll
    for (int c = 0; c < 8; ++c) { float v = row[c] + bb[c]; s += v; sq += v * v; }
  }
  ssum[threadIdx.x] = s; ssq[threadIdx.x] = sq;
  __syncthreads();
  for (int off = 128; off > 0; off >>= 1) {
    if ((int)threadIdx.x < off) {
      ssum[threadIdx.x] += ssum[threadIdx.x + off];
      ssq [threadIdx.x] += ssq [threadIdx.x + off];
    }
    __syncthreads();
  }
  if (threadIdx.x == 0) {
    float mn = ssum[0] / 32768.f;
    float vr = ssq[0] / 32768.f - mn * mn;
    mean[b] = mn;
    rstd[b] = rsqrtf(vr + 1e-5f);
  }
}

__global__ void k_gn_apply(const float* __restrict__ Y, const float* __restrict__ bias,
                           const float* __restrict__ mean, const float* __restrict__ rstd,
                           const float* __restrict__ gam, const float* __restrict__ bet,
                           float* __restrict__ out)
{
  int idx = blockIdx.x * 256 + threadIdx.x;          // over M*256
  int c = idx & 255;
  int m = idx >> 8;
  int n = m >> 12;
  int sg = n * 32 + (c >> 3);
  float v = (Y[idx] + bias[c] - mean[sg]) * rstd[sg] * gam[c] + bet[c];
  out[idx] = fmaxf(v, 0.f);
}

// ---------------------------------------------------------------------------
// InstanceNorm over NHWC [M][Cc] (per n, per channel; pre-norm bias cancels)
// ---------------------------------------------------------------------------
__global__ void k_in_stats(const float* __restrict__ X, int Cc,
                           float* __restrict__ mean, float* __restrict__ rstd)
{
  __shared__ float ssum[256], ssq[256];
  int b = blockIdx.x;                 // N*Cc blocks
  int n = b / Cc, ch = b - n * Cc;
  float s = 0.f, sq = 0.f;
  for (int p = threadIdx.x; p < HWDIM; p += 256) {
    float v = X[((size_t)(n * HWDIM + p)) * Cc + ch];
    s += v; sq += v * v;
  }
  ssum[threadIdx.x] = s; ssq[threadIdx.x] = sq;
  __syncthreads();
  for (int off = 128; off > 0; off >>= 1) {
    if ((int)threadIdx.x < off) {
      ssum[threadIdx.x] += ssum[threadIdx.x + off];
      ssq [threadIdx.x] += ssq [threadIdx.x + off];
    }
    __syncthreads();
  }
  if (threadIdx.x == 0) {
    float mn = ssum[0] / 4096.f;
    float vr = ssq[0] / 4096.f - mn * mn;
    mean[b] = mn;
    rstd[b] = rsqrtf(vr + 1e-5f);
  }
}

// relu((x-m)*rs) -> bf16 at [m][outOff+ch] with row stride outLd
__global__ void k_in_apply_bf16(const float* __restrict__ X, int Cc,
                                const float* __restrict__ mean, const float* __restrict__ rstd,
                                unsigned short* __restrict__ out, int outLd, int outOff)
{
  int idx = blockIdx.x * 256 + threadIdx.x;          // over M*Cc
  int ch = idx % Cc;
  int m  = idx / Cc;
  int n  = m >> 12;
  int s  = n * Cc + ch;
  float v = fmaxf((X[idx] - mean[s]) * rstd[s], 0.f);
  out[(size_t)m * outLd + outOff + ch] = f2bf(v);
}

// relu((x-m)*rs) in place (fp32) — for the ltrb feature used by border_align
__global__ void k_in_apply_f32(float* __restrict__ X, int Cc,
                               const float* __restrict__ mean, const float* __restrict__ rstd)
{
  int idx = blockIdx.x * 256 + threadIdx.x;          // over M*Cc
  int ch = idx % Cc;
  int m  = idx / Cc;
  int n  = m >> 12;
  int s  = n * Cc + ch;
  X[idx] = fmaxf((X[idx] - mean[s]) * rstd[s], 0.f);
}

// ---------------------------------------------------------------------------
// Border align: F NHWC [M][4*cb] fp32, boxes [N*HW][4] -> scatter bf16 into
// AC [M][5*cb] at columns [0, 4*cb) following the reference reshape bijection.
// ---------------------------------------------------------------------------
__global__ void k_border_align(const float* __restrict__ F, const float* __restrict__ boxes,
                               unsigned short* __restrict__ AC, int cb)
{
  int idx = blockIdx.x * 256 + threadIdx.x;          // over N*HW*4*cb
  int c  = idx % cb;
  int r1 = idx / cb;
  int j  = r1 & 3;
  int r2 = r1 >> 2;
  int p  = r2 & 4095;
  int n  = r2 >> 12;
  const int C4 = 4 * cb, C5 = 5 * cb;

  const float* bx = boxes + ((size_t)(n * HWDIM + p)) * 4;
  float x1 = bx[0], y1 = bx[1], x2 = bx[2], y2 = bx[3];
  const float* Fn = F + (size_t)(n * HWDIM) * C4 + j * cb + c;

  float best = -3.4e38f;
  for (int t = 0; t <= 10; ++t) {
    float tt = (float)t / 10.f;
    float xx, yy;
    if      (j == 0) { xx = x1 + tt * (x2 - x1); yy = y1; }                 // top
    else if (j == 1) { xx = x1;                  yy = y1 + tt * (y2 - y1); } // left
    else if (j == 2) { xx = x1 + tt * (x2 - x1); yy = y2; }                 // bottom
    else             { xx = x2;                  yy = y1 + tt * (y2 - y1); } // right
    float x0f = floorf(xx), y0f = floorf(yy);
    float lx = xx - x0f,   ly = yy - y0f;
    int x0 = min(max((int)x0f, 0), 63);
    int y0 = min(max((int)y0f, 0), 63);
    int x1i = min(x0 + 1, 63);
    int y1i = min(y0 + 1, 63);
    const float* r0  = Fn + (size_t)(y0  * 64) * C4;
    const float* r1p = Fn + (size_t)(y1i * 64) * C4;
    float f00 = r0 [(size_t)x0  * C4], f01 = r0 [(size_t)x1i * C4];
    float f10 = r1p[(size_t)x0  * C4], f11 = r1p[(size_t)x1i * C4];
    float v = f00 * (1.f - ly) * (1.f - lx) + f01 * (1.f - ly) * lx
            + f10 * ly * (1.f - lx)        + f11 * ly * lx;
    best = fmaxf(best, v);
  }
  // reference: transpose(la,(0,3,1,2)).reshape(n,4cb,h,w): L = c*HW*4 + p*4 + j
  int L  = c * (HWDIM * 4) + p * 4 + j;
  int ch = L >> 12;          // / HW
  int q  = L & 4095;
  AC[((size_t)(n * HWDIM + q)) * C5 + ch] = f2bf(best);
}

// ---------------------------------------------------------------------------
// Output / post kernels
// ---------------------------------------------------------------------------
// S [M][ldS] (+bias) -> NCHW fp32 [N][Co][H][W]
__global__ void k_add_bias_nchw(const float* __restrict__ S, int ldS, int Co,
                                const float* __restrict__ bias, float* __restrict__ out)
{
  int idx = blockIdx.x * 256 + threadIdx.x;          // over M*Co
  int co = idx % Co;
  int m  = idx / Co;
  int n  = m >> 12, p = m & 4095;
  out[((size_t)n * Co + co) * HWDIM + p] = S[(size_t)m * ldS + co] + bias[co];
}

// relu(G + bias) -> bf16 [M][256]
__global__ void k_bias_relu_bf16(const float* __restrict__ G, const float* __restrict__ bias,
                                 unsigned short* __restrict__ out)
{
  int idx = blockIdx.x * 256 + threadIdx.x;          // over M*256
  int c = idx & 255;
  out[idx] = f2bf(fmaxf(G[idx] + bias[c], 0.f));
}

// bbox_pred / centerness / pre_boxes / align_boxes from packed head out S2 [M][16]
__global__ void k_bbox_post(const float* __restrict__ S2, const float* __restrict__ shifts,
                            const float* __restrict__ bpb, const float* __restrict__ ctrb,
                            const float* __restrict__ scale,
                            float* __restrict__ out_bbox, float* __restrict__ out_ctr,
                            float* __restrict__ out_pre, float* __restrict__ boxes)
{
  int m = blockIdx.x * 256 + threadIdx.x;            // over M
  int n = m >> 12, p = m & 4095;
  float sc = scale[0];
  float v[4];
#pragma unroll
  for (int j = 0; j < 4; ++j) {
    v[j] = fmaxf((S2[(size_t)m * 16 + j] + bpb[j]) * sc, 0.f) * 8.f;
    out_bbox[((size_t)n * 4 + j) * HWDIM + p] = v[j];
  }
  out_ctr[(size_t)n * HWDIM + p] = S2[(size_t)m * 16 + 4] + ctrb[0];
  float sx = shifts[(size_t)m * 2], sy = shifts[(size_t)m * 2 + 1];
  float b0 = sx - v[0], b1 = sy - v[1], b2 = sx + v[2], b3 = sy + v[3];
  out_pre[(size_t)m * 4 + 0] = b0;
  out_pre[(size_t)m * 4 + 1] = b1;
  out_pre[(size_t)m * 4 + 2] = b2;
  out_pre[(size_t)m * 4 + 3] = b3;
  boxes[(size_t)m * 4 + 0] = fminf(fmaxf(b0 * 0.125f, 0.f), 63.f);
  boxes[(size_t)m * 4 + 1] = fminf(fmaxf(b1 * 0.125f, 0.f), 63.f);
  boxes[(size_t)m * 4 + 2] = fminf(fmaxf(b2 * 0.125f, 0.f), 63.f);
  boxes[(size_t)m * 4 + 3] = fminf(fmaxf(b3 * 0.125f, 0.f), 63.f);
}

// ---------------------------------------------------------------------------
// Host orchestration
// ---------------------------------------------------------------------------
static inline int ceilDiv(int a, int b) { return (a + b - 1) / b; }

extern "C" void kernel_launch(void* const* d_in, const int* in_sizes, int n_in,
                              void* d_out, int out_size, void* d_ws, size_t ws_size,
                              hipStream_t stream)
{
  (void)in_sizes; (void)n_in; (void)out_size; (void)ws_size;

  // inputs in setup_inputs() insertion order
  const float* feature      = (const float*)d_in[0];
  const float* shifts       = (const float*)d_in[1];
  const float* cls_w        = (const float*)d_in[2];
  const float* cls_b        = (const float*)d_in[3];
  const float* cls_gn_g     = (const float*)d_in[4];
  const float* cls_gn_b     = (const float*)d_in[5];
  const float* box_w        = (const float*)d_in[6];
  const float* box_b        = (const float*)d_in[7];
  const float* box_gn_g     = (const float*)d_in[8];
  const float* box_gn_b     = (const float*)d_in[9];
  const float* cls_score_w  = (const float*)d_in[10];
  const float* cls_score_b  = (const float*)d_in[11];
  const float* bbox_pred_w  = (const float*)d_in[12];
  const float* bbox_pred_b  = (const float*)d_in[13];
  const float* ctr_w        = (const float*)d_in[14];
  const float* ctr_b        = (const float*)d_in[15];
  const float* scale        = (const float*)d_in[16];
  const float* bcls_cur_w   = (const float*)d_in[17];
  const float* bcls_ltrb_w  = (const float*)d_in[19];
  const float* bcls_out_w   = (const float*)d_in[21];
  const float* bcls_out_b   = (const float*)d_in[22];
  const float* bbx_cur_w    = (const float*)d_in[23];
  const float* bbx_ltrb_w   = (const float*)d_in[25];
  const float* bbx_out_w    = (const float*)d_in[27];
  const float* bbx_out_b    = (const float*)d_in[28];
  const float* bcls_score_w = (const float*)d_in[29];
  const float* bcls_score_b = (const float*)d_in[30];
  const float* bpred2_w     = (const float*)d_in[31];
  const float* bpred2_b     = (const float*)d_in[32];
  // (cur_b/ltrb_b at 18/20/24/26 cancel inside instance_norm)

  // workspace layout (bytes)
  char* ws = (char*)d_ws;
  float*          ACT_CLS = (float*)(ws + 0);                         //  8.4 MB
  float*          ACT_BOX = (float*)(ws + 8388608);                   //  8.4 MB
  unsigned short* COL     = (unsigned short*)(ws + 16777216);         // 37.7 MB
  float*          FBUF    = (float*)(ws + 16777216);                  // aliases COL
  float*          GOUT    = (float*)(ws + 54525952);                  //  8.4 MB
  unsigned short* AC      = (unsigned short*)(ws + 62914560);         // 21.0 MB
  unsigned short* WB      = (unsigned short*)(ws + 83886080);         //  1.2 MB
  float*          S1      = (float*)(ws + 85065728);                  //  2.6 MB
  float*          S2      = (float*)(ws + 87687168);                  //  0.5 MB
  float*          BOXES   = (float*)(ws + 88211456);                  //  128 KB
  float*          MEANB   = (float*)(ws + 88342528);                  //    8 KB
  float*          RSTDB   = (float*)(ws + 88350720);                  //    8 KB
  unsigned short* ABUF    = (unsigned short*)(ws + 88358912);         //  4.2 MB

  // output regions (fp32, concatenated)
  float* o = (float*)d_out;
  float* out_logits  = o;                       // 2*80*4096
  float* out_bbox    = out_logits  + 655360;    // 2*4*4096
  float* out_ctr     = out_bbox    + 32768;     // 2*1*4096
  float* out_blogits = out_ctr     + 8192;      // 2*80*4096
  float* out_breg    = out_blogits + 655360;    // 2*4*4096
  float* out_pre     = out_breg    + 32768;     // 2*4096*4

  const dim3 blk(256), gblk(128);

  // GEMM dispatcher: full 64-wide tiles with NT=4, 16-wide remainders with NT=1
  auto launch_gemm = [&](const unsigned short* A, const unsigned short* Bw, float* Dst,
                         int Nd, int Kd) {
    int nFull = Nd / 64;
    if (nFull > 0)
      k_gemm_bf16<4><<<dim3(nFull, 32), gblk, 0, stream>>>(A, Bw, Dst, Nd, Kd, 0);
    int rem = (Nd / 16) & 3;
    for (int r = 0; r < rem; ++r)
      k_gemm_bf16<1><<<dim3(1, 32), gblk, 0, stream>>>(A, Bw, Dst, Nd, Kd, nFull * 64 + r * 16);
  };

  // ---- subnet (4x conv3x3 + GN + relu) ----
  auto run_subnet = [&](const float* wsrc, const float* bsrc, const float* gsrc,
                        const float* besrc, float* ACT) {
    k_nchw2nhwc<<<MROWS * CDIM / 256, blk, 0, stream>>>(feature, ACT);
    for (int i = 0; i < 4; ++i) {
      k_im2col<<<MROWS * CDIM / 256, blk, 0, stream>>>(ACT, COL);
      k_conv_weight<<<ceilDiv(CDIM * KCONV, 256), blk, 0, stream>>>(
          wsrc + (size_t)i * CDIM * KCONV, WB, CDIM * KCONV);
      launch_gemm(COL, WB, GOUT, CDIM, KCONV);
      k_gn_stats<<<64, blk, 0, stream>>>(GOUT, bsrc + i * CDIM, MEANB, RSTDB);
      k_gn_apply<<<MROWS * CDIM / 256, blk, 0, stream>>>(
          GOUT, bsrc + i * CDIM, MEANB, RSTDB, gsrc + i * CDIM, besrc + i * CDIM, ACT);
    }
  };
  run_subnet(cls_w, cls_b, cls_gn_g, cls_gn_b, ACT_CLS);
  run_subnet(box_w, box_b, box_gn_g, box_gn_b, ACT_BOX);

  // ---- heads (3x3 convs as im2col GEMMs) ----
  k_im2col<<<MROWS * CDIM / 256, blk, 0, stream>>>(ACT_CLS, COL);
  k_conv_weight<<<ceilDiv(80 * KCONV, 256), blk, 0, stream>>>(cls_score_w, WB, 80 * KCONV);
  launch_gemm(COL, WB, S1, 80, KCONV);

  k_im2col<<<MROWS * CDIM / 256, blk, 0, stream>>>(ACT_BOX, COL);
  k_pack_head<<<ceilDiv(16 * KCONV, 256), blk, 0, stream>>>(bbox_pred_w, ctr_w, WB);
  launch_gemm(COL, WB, S2, 16, KCONV);

  k_add_bias_nchw<<<MROWS * 80 / 256, blk, 0, stream>>>(S1, 80, 80, cls_score_b, out_logits);
  k_bbox_post<<<MROWS / 256, blk, 0, stream>>>(S2, shifts, bbox_pred_b, ctr_b, scale,
                                               out_bbox, out_ctr, out_pre, BOXES);

  // ---- border branch (cb = 256 for cls, 128 for box) ----
  auto run_border = [&](const float* SUB, int cb,
                        const float* cur_w, const float* ltrb_w,
                        const float* outw, const float* outb,
                        const float* score_w, const float* score_b,
                        int scoreN, int scoreCo, float* headOut, float* finalOut) {
    const int C4 = 4 * cb, C5 = 5 * cb;
    k_cast_pad<<<MROWS * CDIM / 256, blk, 0, stream>>>(SUB, ABUF, MROWS * CDIM, MROWS * CDIM);
    // cur: 1x1 conv -> IN -> relu -> AC cols [4cb, 5cb)
    k_cast_pad<<<ceilDiv(cb * CDIM, 256), blk, 0, stream>>>(cur_w, WB, cb * CDIM, cb * CDIM);
    launch_gemm(ABUF, WB, GOUT, cb, CDIM);
    k_in_stats<<<NBATCH * cb, blk, 0, stream>>>(GOUT, cb, MEANB, RSTDB);
    k_in_apply_bf16<<<MROWS * cb / 256, blk, 0, stream>>>(GOUT, cb, MEANB, RSTDB, AC, C5, C4);
    // ltrb: 1x1 conv -> IN -> relu (fp32, for sampling)
    k_cast_pad<<<ceilDiv(C4 * CDIM, 256), blk, 0, stream>>>(ltrb_w, WB, C4 * CDIM, C4 * CDIM);
    launch_gemm(ABUF, WB, FBUF, C4, CDIM);
    k_in_stats<<<NBATCH * C4, blk, 0, stream>>>(FBUF, C4, MEANB, RSTDB);
    k_in_apply_f32<<<MROWS * C4 / 256, blk, 0, stream>>>(FBUF, C4, MEANB, RSTDB);
    // border align -> AC cols [0, 4cb)
    k_border_align<<<NBATCH * HWDIM * 4 * cb / 256, blk, 0, stream>>>(FBUF, BOXES, AC, cb);
    // out: 1x1 conv over 5cb -> relu -> bf16
    k_cast_pad<<<ceilDiv(CDIM * C5, 256), blk, 0, stream>>>(outw, WB, CDIM * C5, CDIM * C5);
    launch_gemm(AC, WB, GOUT, CDIM, C5);
    k_bias_relu_bf16<<<MROWS * CDIM / 256, blk, 0, stream>>>(GOUT, outb, ABUF);
    // final 1x1 score conv
    k_cast_pad<<<ceilDiv(scoreN * CDIM, 256), blk, 0, stream>>>(score_w, WB, scoreCo * CDIM, scoreN * CDIM);
    launch_gemm(ABUF, WB, headOut, scoreN, CDIM);
    k_add_bias_nchw<<<MROWS * scoreCo / 256, blk, 0, stream>>>(headOut, scoreN, scoreCo, score_b, finalOut);
  };

  run_border(ACT_CLS, 256, bcls_cur_w, bcls_ltrb_w, bcls_out_w, bcls_out_b,
             bcls_score_w, bcls_score_b, 80, 80, S1, out_blogits);
  run_border(ACT_BOX, 128, bbx_cur_w, bbx_ltrb_w, bbx_out_w, bbx_out_b,
             bpred2_w, bpred2_b, 16, 4, S2, out_breg);
}
